// SparseRelativeMultiHeadAttention_87436944212085
// MI455X (gfx1250) — matrix operand order repaired and checked by reading, not compiled, not run on hardware
//
#include <hip/hip_runtime.h>
#include <cfloat>
#include <cmath>

// ---------------------------------------------------------------------------
// SparseRelativeMultiHeadAttention for MI455X (gfx1250, wave32, WMMA + TDM).
// All heavy math runs through v_wmma_f32_16x16x32_bf16 (fp32 accumulate).
// The relative-position "skew" is folded analytically:
//   rel[i,j] = (q_i . L[valid[j] - i + 1023])  ==  R[i, valid[j]-i+1023],
// and the 16x2048 R strip per query tile is computed by WMMA into LDS (bf16),
// so the 268 MB rel tensor and the 64 MB score tensor never touch HBM.
// GEMM A panels are staged into LDS via the Tensor Data Mover (TENSORcnt).
// ---------------------------------------------------------------------------

#define BB 2
#define T1V 1024
#define T2V 512
#define DM 1024
#define NH 16
#define DH 64
#define SCALEV 0.125f
#define EPSV 1e-5f

typedef __bf16 bf16_t;
typedef __attribute__((ext_vector_type(16))) __bf16 v16bf;
typedef __attribute__((ext_vector_type(8)))  float  v8f;
typedef unsigned int u32x4 __attribute__((ext_vector_type(4)));
typedef int i32x4 __attribute__((ext_vector_type(4)));
typedef int i32x8 __attribute__((ext_vector_type(8)));

struct __align__(16) bf8 { bf16_t x[8]; };

union V16U { v16bf v; bf16_t e[16]; };

__device__ __forceinline__ v16bf pack2(bf8 lo, bf8 hi) {
  V16U u;
#pragma unroll
  for (int i = 0; i < 8; ++i) { u.e[i] = lo.x[i]; u.e[i + 8] = hi.x[i]; }
  return u.v;
}

// A operand (16x32 bf16), per ISA table: lane&15 = row M.
// lanes 0-15 hold K = kk0+{0..7, 16..23}; lanes 16-31 hold K = kk0+{8..15, 24..31}.
__device__ __forceinline__ v16bf load_A(const bf16_t* row_ptr, int kk0, int lane) {
  const int koff = kk0 + ((lane & 16) ? 8 : 0);
  bf8 lo = *(const bf8*)(row_ptr + koff);
  bf8 hi = *(const bf8*)(row_ptr + koff + 16);
  return pack2(lo, hi);
}

// B operand (32x16 bf16), same K order as A: lane&15 = col N.
// lanes 0-15 hold K = kk0+{0..7, 16..23}; lanes 16-31 hold K = kk0+{8..15, 24..31}.
__device__ __forceinline__ v16bf load_B(const bf16_t* col_ptr, int kk0, int lane) {
  const int koff = kk0 + ((lane & 16) ? 8 : 0);
  bf8 lo = *(const bf8*)(col_ptr + koff);
  bf8 hi = *(const bf8*)(col_ptr + koff + 16);
  return pack2(lo, hi);
}

__device__ __forceinline__ v8f wmma_bf16(v16bf a, v16bf b, v8f c) {
  return __builtin_amdgcn_wmma_f32_16x16x32_bf16(false, a, false, b, (short)0, c,
                                                 false, false);
}

// ---------------------------------------------------------------------------
// Tensor Data Mover: 2D tile load Global->LDS. D# built per ISA 08 §8.3/8.4.
// data_size = 2 bytes (bf16). Dimensions/strides in elements. Issue from one
// wave only (TDM ignores EXEC, so guard must be wave-uniform).
// ---------------------------------------------------------------------------
__device__ __forceinline__ void tdm_load_2d_bf16(unsigned lds_byte_off,
                                                 const void* gptr,
                                                 unsigned tensor_d0,
                                                 unsigned tensor_d1,
                                                 unsigned tile_d0,
                                                 unsigned tile_d1,
                                                 unsigned long long d0_stride) {
  const unsigned long long ga = (unsigned long long)(uintptr_t)gptr;
  u32x4 g0;
  g0[0] = 1u;                                   // count=1 (valid), user mode
  g0[1] = lds_byte_off;                         // lds_addr (bytes)
  g0[2] = (unsigned)ga;                         // global_addr[31:0]
  g0[3] = (unsigned)((ga >> 32) & 0x01ffffffu)  // global_addr[56:32]
          | (2u << 30);                         // type = 2 ("image")
  i32x8 g1;
  g1[0] = (int)(1u << 16);                      // data_size=1 -> 2 bytes
  g1[1] = (int)((tensor_d0 & 0xffffu) << 16);   // tensor_dim0[15:0]
  g1[2] = (int)((tensor_d0 >> 16) | ((tensor_d1 & 0xffffu) << 16));
  g1[3] = (int)((tensor_d1 >> 16) | (tile_d0 << 16));   // tile_dim0
  g1[4] = (int)(tile_d1 & 0xffffu);             // tile_dim1 (tile_dim2 = 0)
  g1[5] = (int)(unsigned)(d0_stride & 0xffffffffu);     // dim0_stride[31:0]
  g1[6] = (int)(unsigned)((d0_stride >> 32) & 0xffffu); // dim0_stride[47:32]
  g1[7] = 0;
  i32x4 gz = {0, 0, 0, 0};
#if defined(__clang_major__) && (__clang_major__ >= 23)
  i32x8 gz8 = {0, 0, 0, 0, 0, 0, 0, 0};
  __builtin_amdgcn_tensor_load_to_lds(g0, g1, gz, gz, gz8, 0);
#else
  __builtin_amdgcn_tensor_load_to_lds(g0, g1, gz, gz, 0);
#endif
}

// ---------------------------------------------------------------------------
// Prep kernels
// ---------------------------------------------------------------------------

// Wt[n*K + k] = bf16(W[k*N + n])  (so GEMM contraction is the fast axis)
__global__ void transpose_w_kernel(const float* __restrict__ W,
                                   bf16_t* __restrict__ Wt, int K, int N) {
  __shared__ float tile[32][33];
  const int n0 = blockIdx.x * 32, k0 = blockIdx.y * 32;
  const int tx = threadIdx.x, ty = threadIdx.y;  // block (32, 8)
  for (int r = ty; r < 32; r += 8)
    tile[r][tx] = W[(size_t)(k0 + r) * N + n0 + tx];
  __syncthreads();
  for (int r = ty; r < 32; r += 8)
    Wt[(size_t)(n0 + r) * K + k0 + tx] = (bf16_t)tile[tx][r];
}

// Lb: 2048 x 64 bf16, row 2047 zero-padded.
__global__ void convert_lut_kernel(const float* __restrict__ L,
                                   bf16_t* __restrict__ Lb) {
  const int i = blockIdx.x * 256 + threadIdx.x;  // 2048*64 total
  const int r = i >> 6;
  Lb[i] = (r < 2 * T1V - 1) ? (bf16_t)L[i] : (bf16_t)0.0f;
}

// LayerNorm over last dim (1024), output bf16. One 256-thread block per row.
__global__ void ln_kernel(const float* __restrict__ Q, const float* __restrict__ g,
                          const float* __restrict__ bta, bf16_t* __restrict__ Qn) {
  const int row = blockIdx.x;
  const float* x = Q + (size_t)row * DM;
  const int tid = threadIdx.x;
  float v[4], s = 0.f, sq = 0.f;
#pragma unroll
  for (int i = 0; i < 4; ++i) {
    v[i] = x[tid + 256 * i];
    s += v[i]; sq += v[i] * v[i];
  }
  const int lane = tid & 31, wv = tid >> 5;
#pragma unroll
  for (int off = 16; off >= 1; off >>= 1) {
    s += __shfl_xor(s, off, 32);
    sq += __shfl_xor(sq, off, 32);
  }
  __shared__ float rs[8], rq[8];
  if (lane == 0) { rs[wv] = s; rq[wv] = sq; }
  __syncthreads();
  s = 0.f; sq = 0.f;
#pragma unroll
  for (int i = 0; i < 8; ++i) { s += rs[i]; sq += rq[i]; }
  const float mean = s * (1.f / DM);
  const float var = sq * (1.f / DM) - mean * mean;
  const float rstd = rsqrtf(var + EPSV);
#pragma unroll
  for (int i = 0; i < 4; ++i) {
    const int c = tid + 256 * i;
    Qn[(size_t)row * DM + c] = (bf16_t)((v[i] - mean) * rstd * g[c] + bta[c]);
  }
}

// K/V row gather by valid_frames + bf16 convert. One block per (b, t2).
__global__ void gather_kv_kernel(const float* __restrict__ Kin,
                                 const float* __restrict__ Vin,
                                 const int* __restrict__ vf,
                                 bf16_t* __restrict__ Kg, bf16_t* __restrict__ Vg) {
  const int row = blockIdx.x;  // b*T2 + t2
  const int b = row >> 9, t2 = row & (T2V - 1);
  const int src = vf[t2];
  const float* kp = Kin + ((size_t)b * T1V + src) * DM;
  const float* vp = Vin + ((size_t)b * T1V + src) * DM;
  const int i = threadIdx.x * 4;  // 256 threads x float4 = 1024
  const float4 k4 = *(const float4*)(kp + i);
  const float4 v4 = *(const float4*)(vp + i);
  bf16_t* ko = Kg + (size_t)row * DM + i;
  bf16_t* vo = Vg + (size_t)row * DM + i;
  ko[0] = (bf16_t)k4.x; ko[1] = (bf16_t)k4.y; ko[2] = (bf16_t)k4.z; ko[3] = (bf16_t)k4.w;
  vo[0] = (bf16_t)v4.x; vo[1] = (bf16_t)v4.y; vo[2] = (bf16_t)v4.z; vo[3] = (bf16_t)v4.w;
}

// ---------------------------------------------------------------------------
// WMMA GEMM: C(MxN) = A(Mx1024 bf16 row-major) * Bt(Nx1024 bf16)^T, N=1024.
// Workgroup (4 waves) owns a 16x256 slab: the 16x1024 A panel is TDM-staged
// into LDS once, then each wave computes 4 16x16 tiles (one per N step).
// MODE 0: store bf16 head-major [b,h,t,d]     (q/k projections)
// MODE 1: store bf16 d-major    [b,h,d,t]     (v projection, for PV operand B)
// MODE 2: store f32  out = C + bias[col] + addF[row,col]   (final output GEMM)
// ---------------------------------------------------------------------------
template <int MODE>
__global__ void gemm16_kernel(const bf16_t* __restrict__ A,
                              const bf16_t* __restrict__ Bt, int rpb,
                              bf16_t* __restrict__ outH, float* __restrict__ outF,
                              const float* __restrict__ bias,
                              const float* __restrict__ addF) {
  __shared__ bf16_t Atile[16 * DM];  // 32 KB; sole LDS object -> offset 0
  const int wave = threadIdx.x >> 5;
  const int lane = threadIdx.x & 31;
  const int l15 = lane & 15;
  const int hi8 = (lane & 16) ? 8 : 0;
  const int m0 = blockIdx.x << 4;

  if (wave == 0) {  // wave-uniform guard: TDM ignores EXEC
    tdm_load_2d_bf16(/*lds_off=*/0, A + (size_t)m0 * DM,
                     /*tensor_d0=*/DM, /*tensor_d1=*/16,
                     /*tile_d0=*/DM, /*tile_d1=*/16, /*d0_stride=*/DM);
    __builtin_amdgcn_s_wait_tensorcnt((short)0);
  }
  __syncthreads();

  const bf16_t* arow = Atile + l15 * DM;  // LDS A operand, shared by all waves
#pragma unroll 1
  for (int ny = 0; ny < 4; ++ny) {
    const int n0 = (blockIdx.y << 8) + (ny << 6) + (wave << 4);
    const bf16_t* bcol = Bt + (size_t)(n0 + l15) * DM;
    v8f c = {};
    for (int kk = 0; kk < DM; kk += 32) {
      if ((kk & 127) == 0)  // prefetch streamed B panel ~4 steps ahead
        __builtin_prefetch(bcol + kk + 128, 0, 1);
      c = wmma_bf16(load_A(arow, kk, lane), load_B(bcol, kk, lane), c);
    }
    const int col = n0 + l15;
#pragma unroll
    for (int r = 0; r < 8; ++r) {
      const int row = m0 + hi8 + r;  // C layout: VGPR r -> M = r (+8 hi lanes)
      if constexpr (MODE == 0) {
        const int b = row / rpb, t = row % rpb, h = col >> 6, d = col & 63;
        outH[(((size_t)b * NH + h) * rpb + t) * DH + d] = (bf16_t)c[r];
      } else if constexpr (MODE == 1) {
        const int b = row / rpb, t = row % rpb, h = col >> 6, d = col & 63;
        outH[(((size_t)b * NH + h) * DH + d) * rpb + t] = (bf16_t)c[r];
      } else {
        outF[(size_t)row * DM + col] =
            c[r] + bias[col] + addF[(size_t)row * DM + col];
      }
    }
  }
}

// ---------------------------------------------------------------------------
// Fused attention: one block (4 waves) per (b, h, 16-query tile).
// LDS: R strip 16x2048 bf16 (64KB) -> S 16x512 f32 (32KB) -> P 16x512 bf16
// (P aliases R after the phase-2 barrier). Total dynamic LDS = 96 KB.
// ---------------------------------------------------------------------------
__global__ void attn_kernel(const bf16_t* __restrict__ qh,
                            const bf16_t* __restrict__ kh,
                            const bf16_t* __restrict__ vT,
                            const bf16_t* __restrict__ Lb,
                            const int* __restrict__ vf,
                            const unsigned char* __restrict__ mask,
                            bf16_t* __restrict__ ctxH, float* __restrict__ ctxF) {
  extern __shared__ char smem[];
  bf16_t* Rbuf = (bf16_t*)smem;                     // 16 x 2048 bf16
  float* Sbuf = (float*)(smem + 16 * 2048 * 2);     // 16 x 512 f32
  bf16_t* Pbuf = (bf16_t*)smem;                     // alias over Rbuf

  const int wave = threadIdx.x >> 5;
  const int lane = threadIdx.x & 31;
  const int l15 = lane & 15;
  const int hi8 = (lane & 16) ? 8 : 0;
  const int i0 = blockIdx.x << 4;
  const int h = blockIdx.y;
  const int b = blockIdx.z;
  const int bh = b * NH + h;

  // q tile A operand (K = 64 split into two K=32 steps), registers for all phases
  const bf16_t* qrow = qh + ((size_t)bh * T1V + i0 + l15) * DH;
  const v16bf a0 = load_A(qrow, 0, lane);
  const v16bf a1 = load_A(qrow, 32, lane);

  // Phase 1: R = q_tile @ L^T  (16 x 2048) via WMMA, bf16 into LDS.
  for (int nt = wave; nt < 128; nt += 4) {
    const int c0 = nt << 4;
    const bf16_t* lrow = Lb + (size_t)(c0 + l15) * DH;  // row 2047 is zero pad
    v8f c = {};
    c = wmma_bf16(a0, load_B(lrow, 0, lane), c);
    c = wmma_bf16(a1, load_B(lrow, 32, lane), c);
#pragma unroll
    for (int r = 0; r < 8; ++r)
      Rbuf[(size_t)(hi8 + r) * 2048 + c0 + l15] = (bf16_t)c[r];
  }
  __syncthreads();

  // Phase 2: S = (q k^T + gathered rel) * SCALE, masked -> -FLT_MAX.
  for (int jt = wave; jt < 32; jt += 4) {
    const int j0 = jt << 4;
    const bf16_t* krow = kh + ((size_t)bh * T2V + j0 + l15) * DH;
    v8f c = {};
    c = wmma_bf16(a0, load_B(krow, 0, lane), c);
    c = wmma_bf16(a1, load_B(krow, 32, lane), c);
    const int vj = vf[j0 + l15];
    const unsigned char* mcol =
        mask + ((size_t)bh * T1V + i0) * T2V + j0 + l15;
#pragma unroll
    for (int r = 0; r < 8; ++r) {
      const int m = hi8 + r;
      const int idx = vj - (i0 + m) + (T1V - 1);  // in [0, 2046]
      float s = (c[r] + (float)Rbuf[(size_t)m * 2048 + idx]) * SCALEV;
      if (mcol[(size_t)m * T2V]) s = -FLT_MAX;
      Sbuf[m * T2V + j0 + l15] = s;
    }
  }
  __syncthreads();

  // Phase 3: row-wise softmax (wave32 shfl reductions), bf16 probs into Pbuf.
  for (int row = wave; row < 16; row += 4) {
    float vals[16];
    float mx = -FLT_MAX;
#pragma unroll
    for (int t = 0; t < 16; ++t) {
      vals[t] = Sbuf[row * T2V + lane + 32 * t];
      mx = fmaxf(mx, vals[t]);
    }
#pragma unroll
    for (int off = 16; off >= 1; off >>= 1) mx = fmaxf(mx, __shfl_xor(mx, off, 32));
    float sum = 0.f;
#pragma unroll
    for (int t = 0; t < 16; ++t) { vals[t] = __expf(vals[t] - mx); sum += vals[t]; }
#pragma unroll
    for (int off = 16; off >= 1; off >>= 1) sum += __shfl_xor(sum, off, 32);
    const float inv = 1.f / sum;
#pragma unroll
    for (int t = 0; t < 16; ++t)
      Pbuf[row * T2V + lane + 32 * t] = (bf16_t)(vals[t] * inv);
  }
  __syncthreads();

  // Phase 4: O = P @ V. Each wave owns one 16-wide d tile; K loop over 512.
  const int n0 = wave << 4;
  const bf16_t* vrow = vT + ((size_t)bh * DH + n0 + l15) * T2V;
  const bf16_t* prow = Pbuf + l15 * T2V;  // LDS A operand
  v8f o = {};
  for (int kk = 0; kk < T2V; kk += 32)
    o = wmma_bf16(load_A(prow, kk, lane), load_B(vrow, kk, lane), o);
#pragma unroll
  for (int r = 0; r < 8; ++r) {
    const int m = hi8 + r;
    const size_t oidx = ((size_t)b * T1V + i0 + m) * DM + h * DH + n0 + l15;
    ctxF[oidx] = o[r];           // fp32 copy for the "+ ctx" epilogue
    ctxH[oidx] = (bf16_t)o[r];   // bf16 copy as A operand of the output GEMM
  }
}

// ---------------------------------------------------------------------------

extern "C" void kernel_launch(void* const* d_in, const int* in_sizes, int n_in,
                              void* d_out, int out_size, void* d_ws, size_t ws_size,
                              hipStream_t stream) {
  const float* Q = (const float*)d_in[0];
  const float* K = (const float*)d_in[1];
  const float* V = (const float*)d_in[2];
  const float* L = (const float*)d_in[3];
  const int* vf = (const int*)d_in[4];
  const unsigned char* mask = (const unsigned char*)d_in[5];
  const float* Wq = (const float*)d_in[6];
  const float* Wk = (const float*)d_in[7];
  const float* Wv = (const float*)d_in[8];
  const float* Wo = (const float*)d_in[9];
  const float* bo = (const float*)d_in[10];
  const float* lg = (const float*)d_in[11];
  const float* lb = (const float*)d_in[12];
  float* out = (float*)d_out;

  // Workspace carve-up (~38 MB), recomputed identically every call.
  char* p = (char*)d_ws;
  auto take = [&](size_t bytes) -> char* {
    char* r = p;
    p += (bytes + 255) & ~(size_t)255;
    return r;
  };
  bf16_t* Wqt = (bf16_t*)take((size_t)DM * DM * 2);
  bf16_t* Wkt = (bf16_t*)take((size_t)DM * DM * 2);
  bf16_t* Wvt = (bf16_t*)take((size_t)DM * DM * 2);
  bf16_t* Wot = (bf16_t*)take((size_t)DM * DM * 2);
  bf16_t* Lb  = (bf16_t*)take((size_t)2048 * DH * 2);
  bf16_t* Qn  = (bf16_t*)take((size_t)BB * T1V * DM * 2);
  bf16_t* Kg  = (bf16_t*)take((size_t)BB * T2V * DM * 2);
  bf16_t* Vg  = (bf16_t*)take((size_t)BB * T2V * DM * 2);
  bf16_t* qhB = (bf16_t*)take((size_t)BB * NH * T1V * DH * 2);
  bf16_t* khB = (bf16_t*)take((size_t)BB * NH * T2V * DH * 2);
  bf16_t* vTB = (bf16_t*)take((size_t)BB * NH * DH * T2V * 2);
  bf16_t* ctxH = (bf16_t*)take((size_t)BB * T1V * DM * 2);
  float*  ctxF = (float*)take((size_t)BB * T1V * DM * 4);

  // 1) weight transpose + bf16 convert, lookup table convert (zero-padded)
  {
    dim3 tb(32, 8), tg(DM / 32, DM / 32);
    transpose_w_kernel<<<tg, tb, 0, stream>>>(Wq, Wqt, DM, DM);
    transpose_w_kernel<<<tg, tb, 0, stream>>>(Wk, Wkt, DM, DM);
    transpose_w_kernel<<<tg, tb, 0, stream>>>(Wv, Wvt, DM, DM);
    transpose_w_kernel<<<tg, tb, 0, stream>>>(Wo, Wot, DM, DM);
  }
  convert_lut_kernel<<<(2048 * DH) / 256, 256, 0, stream>>>(L, Lb);

  // 2) LayerNorm(Q), 3) gather K/V by valid_frames
  ln_kernel<<<BB * T1V, 256, 0, stream>>>(Q, lg, lb, Qn);
  gather_kv_kernel<<<BB * T2V, 256, 0, stream>>>(K, V, vf, Kg, Vg);

  // 4) projections (WMMA, TDM-staged A panels)
  gemm16_kernel<0><<<dim3(BB * T1V / 16, DM / 256), 128, 0, stream>>>(
      Qn, Wqt, T1V, qhB, nullptr, nullptr, nullptr);
  gemm16_kernel<0><<<dim3(BB * T2V / 16, DM / 256), 128, 0, stream>>>(
      Kg, Wkt, T2V, khB, nullptr, nullptr, nullptr);
  gemm16_kernel<1><<<dim3(BB * T2V / 16, DM / 256), 128, 0, stream>>>(
      Vg, Wvt, T2V, vTB, nullptr, nullptr, nullptr);

  // 5) fused rel-skew attention (96 KB dynamic LDS per workgroup)
  attn_kernel<<<dim3(T1V / 16, NH, BB), 128, 96 * 1024, stream>>>(
      qhB, khB, vTB, Lb, vf, mask, ctxH, ctxF);

  // 6) out = ctx @ Wo + bo + ctx (WMMA with fp32 epilogue)
  gemm16_kernel<2><<<dim3(BB * T1V / 16, DM / 256), 128, 0, stream>>>(
      ctxH, Wot, T1V, nullptr, out, bo, ctxF);
}
